// TGL_Fusion_19533511262429
// MI455X (gfx1250) — matrix operand order repaired and checked by reading, not compiled, not run on hardware
//
#include <hip/hip_runtime.h>

// ---------------------------------------------------------------------------
// MI455X (gfx1250):
//  - All GEMMs: v_wmma_f32_16x16x32_bf16 (wave32, fp32 accum).
//  - Attention fused flash-style; logits = (xt - 0.1*zt)^T @ zt.
//  - All global->LDS staging is transpose-free row copies so it can use
//    GLOBAL_LOAD_ASYNC_TO_LDS_B128 (ASYNCcnt) with double buffering.
//  - Layouts: activations [B][n][c] (c contiguous), xt/zt/q [B][n][c],
//    zfg [B][c][n], cat [B][n][512]. Spatial padded 961 -> SN=1024.
// ---------------------------------------------------------------------------

typedef __bf16 bf16;
typedef __attribute__((ext_vector_type(16))) __bf16 v16bf;
typedef __attribute__((ext_vector_type(8)))  __bf16 v8bf;
typedef __attribute__((ext_vector_type(8)))  float  v8f;
typedef __attribute__((ext_vector_type(4)))  int    v4i;

typedef __attribute__((address_space(1))) v4i glb_v4i;
typedef __attribute__((address_space(3))) v4i lds_v4i;

#define SN    1024
#define NREAL 961
#define CCH   256
#define BATCH 32
#define EPSBN 1e-5f

#if defined(__has_builtin)
#if __has_builtin(__builtin_amdgcn_global_load_async_to_lds_b128)
#define HAVE_ASYNC_LDS 1
#endif
#endif
#ifndef HAVE_ASYNC_LDS
#define HAVE_ASYNC_LDS 0
#endif

union V16U { v16bf v; v8bf h[2]; };

// 16B global -> LDS copy; async DMA when available, register path otherwise.
__device__ __forceinline__ void async_copy16(const bf16* g, bf16* l) {
#if HAVE_ASYNC_LDS
  __builtin_amdgcn_global_load_async_to_lds_b128(
      (glb_v4i*)g, (lds_v4i*)l, 0, 0);
#else
  *(v8bf*)l = *(const v8bf*)g;
#endif
}

__device__ __forceinline__ void async_wait0() {
#if HAVE_ASYNC_LDS
#if __has_builtin(__builtin_amdgcn_s_wait_asynccnt)
  __builtin_amdgcn_s_wait_asynccnt(0);
#else
  asm volatile("s_wait_asynccnt 0" ::: "memory");
#endif
#endif
}

__device__ __forceinline__ void wait_ds0() {
#if defined(__has_builtin) && __has_builtin(__builtin_amdgcn_s_wait_dscnt)
  __builtin_amdgcn_s_wait_dscnt(0);
#else
  asm volatile("s_wait_dscnt 0" ::: "memory");
#endif
}

// A fragment: 16(M) x 32(K), LDS row-major [m][k].
// Lanes 0-15/16-31 hold M=lane&15; elems 0..7 <- K=hi*8+e, 8..15 <- K=16+hi*8+(e-8).
__device__ __forceinline__ v16bf load_a_frag(const bf16* base, int ld) {
  const int lane = threadIdx.x & 31;
  const int hi = lane >> 4;
  const bf16* row = base + (lane & 15) * ld;
  V16U u;
  u.h[0] = *(const v8bf*)(row + hi * 8);
  u.h[1] = *(const v8bf*)(row + 16 + hi * 8);
  return u.v;
}

// B fragment: 32(K) x 16(N), staged transposed in LDS as [n][k].
// N=lane&15; elems 0..7 <- K=hi*8+e, 8..15 <- K=16+hi*8+(e-8).
__device__ __forceinline__ v16bf load_b_frag(const bf16* base, int ld) {
  const int lane = threadIdx.x & 31;
  const int hi = lane >> 4;
  const bf16* row = base + (lane & 15) * ld;
  V16U u;
  u.h[0] = *(const v8bf*)(row + hi * 8);
  u.h[1] = *(const v8bf*)(row + 16 + hi * 8);
  return u.v;
}

__device__ __forceinline__ v8f wmma_bf16(v16bf a, v16bf b, v8f c) {
  return __builtin_amdgcn_wmma_f32_16x16x32_bf16(false, a, false, b,
                                                 (short)0, c, false, false);
}

// ---------------------------------------------------------------------------
// fp32 [B][c][961] -> bf16 [B][n][c]   (transpose + pad rows n>=961 with 0)
// ---------------------------------------------------------------------------
__global__ void cvt_act_t(const float* __restrict__ src, bf16* __restrict__ dst) {
  long idx = (long)blockIdx.x * blockDim.x + threadIdx.x;  // B*SN*CCH
  int  c = (int)(idx & (CCH - 1));
  long t = idx >> 8;
  int  n = (int)(t & (SN - 1));
  long b = t >> 10;
  dst[idx] = (n < NREAL) ? (bf16)src[(b * CCH + c) * NREAL + n] : (bf16)0.0f;
}

__global__ void cvt_w(const float* __restrict__ src, bf16* __restrict__ dst, int n) {
  int i = blockIdx.x * blockDim.x + threadIdx.x;
  if (i < n) dst[i] = (bf16)src[i];
}

// q[b][n][c] = xt[b][n][c] - 0.1*zt[b][n][c]   (same layout, pure elementwise)
__global__ void make_q(const bf16* __restrict__ xt, const bf16* __restrict__ zt,
                       bf16* __restrict__ qT) {
  long idx = (long)blockIdx.x * blockDim.x + threadIdx.x;  // B*SN*CCH
  qT[idx] = (bf16)((float)xt[idx] - 0.1f * (float)zt[idx]);
}

// ---------------------------------------------------------------------------
// 1x1-conv GEMM:  Y[b] = W(256 x K) @ X[b]^T + bias [+BN+ReLU]
//   X layout [B][SN][K] (channel-contiguous) -> transpose-free B staging.
//   Double-buffered async LDS staging, 64x64 tile, 8 waves.
// ---------------------------------------------------------------------------
template <bool BNRELU, bool OUTF32, bool TRANSOUT>
__global__ void __launch_bounds__(256) conv_gemm(
    const bf16* __restrict__ W, const bf16* __restrict__ X,
    const float* __restrict__ bias,
    const float* __restrict__ gamma, const float* __restrict__ beta,
    const float* __restrict__ mean,  const float* __restrict__ var,
    bf16* __restrict__ Ybf, float* __restrict__ Yf32,
    int K, long xBatchStride, long yBatchStride, int ldY) {
  __shared__ alignas(16) bf16 sA[2][64 * 40];   // W tile  [m][k]
  __shared__ alignas(16) bf16 sB[2][64 * 40];   // X tile  [n][k]

  const int tid = threadIdx.x;
  const int lane = tid & 31, wave = tid >> 5;
  const int hi = lane >> 4, l15 = lane & 15;
  const int b  = blockIdx.z;
  const int m0 = blockIdx.y * 64;
  const int n0 = blockIdx.x * 64;
  const int wm = wave & 3;
  const int wn = (wave >> 2) * 32;
  const bf16* Xb = X + (long)b * xBatchStride;

  const int rowT = tid >> 2, k8 = (tid & 3) * 8;   // one 16B chunk per thread

  auto issue = [&](int kk, int buf) {
    async_copy16(W  + (long)(m0 + rowT) * K + kk + k8, &sA[buf][rowT * 40 + k8]);
    async_copy16(Xb + (long)(n0 + rowT) * K + kk + k8, &sB[buf][rowT * 40 + k8]);
  };

  v8f acc0 = {0.f, 0.f, 0.f, 0.f, 0.f, 0.f, 0.f, 0.f};
  v8f acc1 = acc0;

  issue(0, 0);
  const int nIter = K >> 5;
  for (int i = 0; i < nIter; ++i) {
    const int cur = i & 1;
    async_wait0();
    __syncthreads();
    if (i + 1 < nIter) issue((i + 1) << 5, 1 - cur);
    v16bf a  = load_a_frag(&sA[cur][wm * 16 * 40], 40);
    v16bf b0 = load_b_frag(&sB[cur][wn * 40], 40);
    v16bf b1 = load_b_frag(&sB[cur][(wn + 16) * 40], 40);
    acc0 = wmma_bf16(a, b0, acc0);
    acc1 = wmma_bf16(a, b1, acc1);
    wait_ds0();   // ds reads done before anyone can recycle this buffer
  }

#pragma unroll
  for (int t = 0; t < 2; ++t) {
    v8f acc = t ? acc1 : acc0;
    int n = n0 + wn + t * 16 + l15;
#pragma unroll
    for (int r = 0; r < 8; ++r) {
      int m = m0 + wm * 16 + r + 8 * hi;
      float v = acc[r] + bias[m];
      if constexpr (BNRELU) {
        float sc = gamma[m] * rsqrtf(var[m] + EPSBN);
        v = fmaxf(v * sc + (beta[m] - mean[m] * sc), 0.f);
      }
      if constexpr (OUTF32) {
        if (n < NREAL)
          Yf32[(long)b * yBatchStride + (long)m * NREAL + n] = v;
      } else if constexpr (TRANSOUT) {
        Ybf[(long)b * yBatchStride + (long)n * ldY + m] = (bf16)v;
      } else {
        Ybf[(long)b * yBatchStride + (long)m * SN + n] = (bf16)v;
      }
    }
  }
}

// ---------------------------------------------------------------------------
// Flash attention: block = 128 query rows, 8 waves, each wave owns 16 rows.
//   S(16x64) = Q(16x256)@K(256x64); online softmax; O(16x256) += P@V.
// K/V tiles double-buffered through the async path. Cols m>=961 masked.
// Output -> cat[b][n][0..256).
// ---------------------------------------------------------------------------
__global__ void __launch_bounds__(256) flash_attn(
    const bf16* __restrict__ Qt,    // q  [B][SN][256]
    const bf16* __restrict__ Kt,    // zt [B][SN][256]  (n-major == K^T)
    const bf16* __restrict__ Vz,    // zfg [B][256][SN]
    bf16* __restrict__ Out) {       // cat [B][SN][512]
  __shared__ alignas(16) bf16 sQ[128 * 264];
  __shared__ alignas(16) bf16 sK[2][64 * 264];    // [m][c]
  __shared__ alignas(16) bf16 sV[2][256 * 72];    // [c][m]
  __shared__ alignas(16) bf16 sP[8 * 16 * 72];    // per-wave P [row][m]

  const int tid = threadIdx.x, lane = tid & 31, wave = tid >> 5;
  const int hi = lane >> 4, l15 = lane & 15;
  const int b  = blockIdx.y;
  const int n0 = blockIdx.x * 128;
  const bf16* Qb = Qt + (long)b * SN * CCH;
  const bf16* Kb = Kt + (long)b * SN * CCH;
  const bf16* Vb = Vz + (long)b * CCH * SN;

  auto issueKV = [&](int j, int buf) {
    const int m0 = j * 64;
#pragma unroll
    for (int it = 0; it < 8; ++it) {
      int idx = it * 256 + tid;
      int r  = idx >> 5, c8 = (idx & 31) * 8;   // K tile: 64 rows x 256 c
      async_copy16(Kb + (long)(m0 + r) * CCH + c8, &sK[buf][r * 264 + c8]);
      int c  = idx >> 3, m8 = (idx & 7) * 8;    // V tile: 256 rows x 64 m
      async_copy16(Vb + (long)c * SN + m0 + m8, &sV[buf][c * 72 + m8]);
    }
  };

#pragma unroll
  for (int it = 0; it < 16; ++it) {             // Q block: 128 rows x 256 c
    int idx = it * 256 + tid;
    int row = idx >> 5, c8 = (idx & 31) * 8;
    async_copy16(Qb + (long)(n0 + row) * CCH + c8, sQ + row * 264 + c8);
  }
  issueKV(0, 0);

  v8f accO[16];
  v8f zero = {0.f, 0.f, 0.f, 0.f, 0.f, 0.f, 0.f, 0.f};
#pragma unroll
  for (int i = 0; i < 16; ++i) accO[i] = zero;
  float rmax[8], rsum[8];
#pragma unroll
  for (int r = 0; r < 8; ++r) { rmax[r] = -1e30f; rsum[r] = 0.f; }

  bf16* sPw = sP + wave * 16 * 72;
  const bf16* sQw = sQ + wave * 16 * 264;

  for (int j = 0; j < 16; ++j) {
    const int m0 = j * 64;
    const int cur = j & 1;
    async_wait0();
    __syncthreads();
    if (j + 1 < 16) issueKV(j + 1, 1 - cur);

    v8f s0 = zero, s1 = zero, s2 = zero, s3 = zero;
#pragma unroll
    for (int kk = 0; kk < 256; kk += 32) {
      v16bf a = load_a_frag(sQw + kk, 264);
      s0 = wmma_bf16(a, load_b_frag(&sK[cur][0 * 16 * 264 + kk], 264), s0);
      s1 = wmma_bf16(a, load_b_frag(&sK[cur][1 * 16 * 264 + kk], 264), s1);
      s2 = wmma_bf16(a, load_b_frag(&sK[cur][2 * 16 * 264 + kk], 264), s2);
      s3 = wmma_bf16(a, load_b_frag(&sK[cur][3 * 16 * 264 + kk], 264), s3);
    }
    v8f sv[4] = {s0, s1, s2, s3};
#pragma unroll
    for (int t = 0; t < 4; ++t) {
      int m = m0 + t * 16 + l15;
      if (m >= NREAL) {
#pragma unroll
        for (int r = 0; r < 8; ++r) sv[t][r] = -1e30f;
      }
    }
    // online softmax; element r <-> row (r+8*hi); cols live across 16 lanes.
#pragma unroll
    for (int r = 0; r < 8; ++r) {
      float mx = fmaxf(fmaxf(sv[0][r], sv[1][r]), fmaxf(sv[2][r], sv[3][r]));
#pragma unroll
      for (int off = 1; off < 16; off <<= 1)
        mx = fmaxf(mx, __shfl_xor(mx, off, 32));
      float nmax = fmaxf(rmax[r], mx);
      float corr = __expf(rmax[r] - nmax);
      rmax[r] = nmax;
      float psum = 0.f;
#pragma unroll
      for (int t = 0; t < 4; ++t) {
        float p = __expf(sv[t][r] - nmax);
        psum += p;
        sPw[(r + 8 * hi) * 72 + t * 16 + l15] = (bf16)p;
      }
#pragma unroll
      for (int off = 1; off < 16; off <<= 1)
        psum += __shfl_xor(psum, off, 32);
      rsum[r] = rsum[r] * corr + psum;
#pragma unroll
      for (int ct = 0; ct < 16; ++ct) accO[ct][r] *= corr;
    }
    // O += P @ V
#pragma unroll
    for (int ks = 0; ks < 64; ks += 32) {
      v16bf a = load_a_frag(sPw + ks, 72);
#pragma unroll
      for (int ct = 0; ct < 16; ++ct) {
        v16bf bv = load_b_frag(&sV[cur][ct * 16 * 72 + ks], 72);
        accO[ct] = wmma_bf16(a, bv, accO[ct]);
      }
    }
    wait_ds0();   // all ds reads of this buffer done before it is recycled
  }

#pragma unroll
  for (int ct = 0; ct < 16; ++ct) {
    int c = ct * 16 + l15;
#pragma unroll
    for (int r = 0; r < 8; ++r) {
      int n = n0 + wave * 16 + r + 8 * hi;
      if (n < NREAL) {
        float v = accO[ct][r] / rsum[r];
        Out[(long)b * SN * 512 + (long)n * 512 + c] = (bf16)v;
      }
    }
  }
}

// ---------------------------------------------------------------------------
extern "C" void kernel_launch(void* const* d_in, const int* in_sizes, int n_in,
                              void* d_out, int out_size, void* d_ws, size_t ws_size,
                              hipStream_t stream) {
  const float* zf  = (const float*)d_in[0];
  const float* xf  = (const float*)d_in[1];
  const float* wq  = (const float*)d_in[2];
  const float* bq  = (const float*)d_in[3];
  const float* wsm = (const float*)d_in[4];
  const float* bs  = (const float*)d_in[5];
  const float* wgx = (const float*)d_in[6];
  const float* bgx = (const float*)d_in[7];
  const float* gxg = (const float*)d_in[8];
  const float* gxb = (const float*)d_in[9];
  const float* gxm = (const float*)d_in[10];
  const float* gxv = (const float*)d_in[11];
  const float* wgz = (const float*)d_in[12];
  const float* bgz = (const float*)d_in[13];
  const float* gzg = (const float*)d_in[14];
  const float* gzb = (const float*)d_in[15];
  const float* gzm = (const float*)d_in[16];
  const float* gzv = (const float*)d_in[17];
  const float* wfi = (const float*)d_in[18];
  const float* bfi = (const float*)d_in[19];
  const float* fig = (const float*)d_in[20];
  const float* fib = (const float*)d_in[21];
  const float* fim = (const float*)d_in[22];
  const float* fiv = (const float*)d_in[23];

  char* wsb = (char*)d_ws;
  size_t off = 0;
  auto take = [&](size_t bytes) -> char* {
    char* p = wsb + off;
    off += (bytes + 255) & ~(size_t)255;
    return p;
  };
  const size_t ACT = (size_t)BATCH * CCH * SN * sizeof(bf16);  // 16.8 MB
  bf16* zfT  = (bf16*)take(ACT);                 // [B][n][c]
  bf16* xfT  = (bf16*)take(ACT);                 // [B][n][c]
  bf16* xtT  = (bf16*)take(ACT);                 // [B][n][c]
  bf16* ztT  = (bf16*)take(ACT);                 // [B][n][c]
  bf16* zfg  = (bf16*)take(ACT);                 // [B][c][n]
  bf16* qT   = (bf16*)take(ACT);                 // [B][n][c]
  bf16* cat  = (bf16*)take(2 * ACT);             // [B][n][512]
  bf16* wq_bf  = (bf16*)take((size_t)CCH * CCH * sizeof(bf16));
  bf16* ws_bf  = (bf16*)take((size_t)CCH * CCH * sizeof(bf16));
  bf16* wgx_bf = (bf16*)take((size_t)CCH * CCH * sizeof(bf16));
  bf16* wgz_bf = (bf16*)take((size_t)CCH * CCH * sizeof(bf16));
  bf16* wfi_bf = (bf16*)take((size_t)CCH * 2 * CCH * sizeof(bf16));
  // total workspace ~135 MB

  const long actElems = (long)BATCH * CCH * SN;
  cvt_act_t<<<(unsigned)(actElems / 256), 256, 0, stream>>>(zf, zfT);
  cvt_act_t<<<(unsigned)(actElems / 256), 256, 0, stream>>>(xf, xfT);
  cvt_w<<<(CCH * CCH + 255) / 256, 256, 0, stream>>>(wq, wq_bf, CCH * CCH);
  cvt_w<<<(CCH * CCH + 255) / 256, 256, 0, stream>>>(wsm, ws_bf, CCH * CCH);
  cvt_w<<<(CCH * CCH + 255) / 256, 256, 0, stream>>>(wgx, wgx_bf, CCH * CCH);
  cvt_w<<<(CCH * CCH + 255) / 256, 256, 0, stream>>>(wgz, wgz_bf, CCH * CCH);
  cvt_w<<<(2 * CCH * CCH + 255) / 256, 256, 0, stream>>>(wfi, wfi_bf, 2 * CCH * CCH);

  dim3 gconv(SN / 64, CCH / 64, BATCH);
  const long actStride = (long)CCH * SN;
  // xt^T = (wq @ xf)^T + bq    -> [B][n][c]
  conv_gemm<false, false, true><<<gconv, 256, 0, stream>>>(
      wq_bf, xfT, bq, nullptr, nullptr, nullptr, nullptr,
      xtT, nullptr, CCH, actStride, actStride, CCH);
  // zt^T = (ws @ zf)^T + bs    -> [B][n][c]
  conv_gemm<false, false, true><<<gconv, 256, 0, stream>>>(
      ws_bf, zfT, bs, nullptr, nullptr, nullptr, nullptr,
      ztT, nullptr, CCH, actStride, actStride, CCH);
  // zfg = bn_relu(wgz @ zf + bgz)  -> [B][c][n]
  conv_gemm<true, false, false><<<gconv, 256, 0, stream>>>(
      wgz_bf, zfT, bgz, gzg, gzb, gzm, gzv,
      zfg, nullptr, CCH, actStride, actStride, SN);
  // xfg -> cat[b][n][256..512)
  conv_gemm<true, false, true><<<gconv, 256, 0, stream>>>(
      wgx_bf, xfT, bgx, gxg, gxb, gxm, gxv,
      cat + CCH, nullptr, CCH, actStride, (long)SN * 2 * CCH, 2 * CCH);

  make_q<<<(unsigned)(actElems / 256), 256, 0, stream>>>(xtT, ztT, qT);

  flash_attn<<<dim3(SN / 128, BATCH), 256, 0, stream>>>(qT, ztT, zfg, cat);

  // out = bn_relu(wfi @ cat + bfi), fp32 [B][256][961]
  conv_gemm<true, true, false><<<gconv, 256, 0, stream>>>(
      wfi_bf, cat, bfi, fig, fib, fim, fiv,
      nullptr, (float*)d_out, 2 * CCH, (long)SN * 2 * CCH, (long)CCH * NREAL, 0);
}
